// FeatureMatchingLoss_50208167690788
// MI455X (gfx1250) — matrix operand order repaired and checked by reading, not compiled, not run on hardware
//
#include <hip/hip_runtime.h>
#include <hip/hip_bf16.h>
#include <math.h>

// Problem constants (B, C, H, W) = (4, 128, 64, 64)
#define BATCH 4
#define CHAN 128
#define HW 4096            // 64*64
#define TAU_INV 10.0f
#define SHIFT 10.0f        // |sim| <= 1/tau = 10 (unit-vector dot / tau)
#define EPS_NORM 1e-12f
#define EPS_DEN 1e-6f

#define NCHUNK 4                   // n-range split per column block
#define NROWS (HW / NCHUNK)        // 1024 rows per WG
#define NTILES (NROWS / 16)        // 64 tiles of 16 rows
#define LDS_PITCH 136              // halfs per LDS row (272B, conflict-avoiding)

typedef _Float16 v16h __attribute__((ext_vector_type(16)));
typedef _Float16 v8h  __attribute__((ext_vector_type(8)));
typedef float    v8f  __attribute__((ext_vector_type(8)));
typedef int      v4i_vs __attribute__((vector_size(16)));   // builtin's pointee type

// CDNA5 async global->LDS copy (ASYNCcnt-tracked, no VGPR round trip).
#if defined(__has_builtin)
#if __has_builtin(__builtin_amdgcn_global_load_async_to_lds_b128)
#define HAVE_ASYNC_LDS 1
#endif
#endif
#ifndef HAVE_ASYNC_LDS
#define HAVE_ASYNC_LDS 0
#endif

#if HAVE_ASYNC_LDS
__device__ __forceinline__ void async_fill(const _Float16* g, _Float16* s) {
    __builtin_amdgcn_global_load_async_to_lds_b128(
        (v4i_vs*)g, (v4i_vs*)s, /*offset=*/0, /*cpol=*/0);
}
#endif

__device__ __forceinline__ void wait_fill() {
#if HAVE_ASYNC_LDS
#if __has_builtin(__builtin_amdgcn_s_wait_asynccnt)
    __builtin_amdgcn_s_wait_asynccnt(0);
#else
    asm volatile("s_wait_asynccnt 0x0" ::: "memory");
#endif
#endif
}

// ---------------------------------------------------------------------------
// Kernel Z: zero per-column sum-exp array and the two scalar accumulators
// ---------------------------------------------------------------------------
__global__ void fml_zero(float* __restrict__ sumexp, float* __restrict__ accum) {
    int t = blockIdx.x * blockDim.x + threadIdx.x;
    if (t < BATCH * HW) sumexp[t] = 0.0f;
    if (t < 2) accum[t] = 0.0f;
}

// ---------------------------------------------------------------------------
// Kernel N: L2-normalize along C, write transposed packed f16 layouts:
//   An[b][pos][c] = fA / max(||fA||, eps)
//   Bn[b][pos][c] = fB / max(||fB||, eps) * (1/tau)     (tau folded into B)
// ---------------------------------------------------------------------------
__global__ void fml_normalize(const float* __restrict__ featA,
                              const float* __restrict__ featB,
                              _Float16* __restrict__ An,
                              _Float16* __restrict__ Bn) {
    int p = blockIdx.x * blockDim.x + threadIdx.x;     // 0 .. BATCH*HW-1
    int b = p >> 12;
    int n = p & (HW - 1);
    const float* fa = featA + (size_t)b * CHAN * HW + n;
    const float* fb = featB + (size_t)b * CHAN * HW + n;

    float sa = 0.0f, sb = 0.0f;
    #pragma unroll 8
    for (int c = 0; c < CHAN; ++c) {
        float va = fa[(size_t)c * HW];
        float vb = fb[(size_t)c * HW];
        sa += va * va;
        sb += vb * vb;
    }
    float ra = 1.0f / fmaxf(sqrtf(sa), EPS_NORM);
    float rb = TAU_INV / fmaxf(sqrtf(sb), EPS_NORM);

    _Float16* oa = An + (size_t)p * CHAN;
    _Float16* ob = Bn + (size_t)p * CHAN;
    #pragma unroll 8
    for (int c = 0; c < CHAN; ++c) {
        oa[c] = (_Float16)(fa[(size_t)c * HW] * ra);
        ob[c] = (_Float16)(fb[(size_t)c * HW] * rb);
    }
}

// B fragment (32x16 f16 KxN, contiguous k order, same as A): lane L<16 -> col N=L, K=k0+0..15;
// lane L>=16 -> col N=L-16, K=k0+16..31. Contiguous per lane in [m][c] layout.
__device__ __forceinline__ v16h load_fragB(const _Float16* __restrict__ colptr,
                                           int kbase, int lane) {
    int koff = (lane & 16) ? 16 : 0;
    v8h b0 = *(const v8h*)(colptr + kbase + koff);
    v8h b1 = *(const v8h*)(colptr + kbase + koff + 8);
    return __builtin_shufflevector(b0, b1,
        0, 1, 2, 3, 4, 5, 6, 7, 8, 9, 10, 11, 12, 13, 14, 15);
}

// A fragment (16x32 f16 MxK) read from the LDS-staged tile.
// lane L<16 -> row M=L, K=k0+0..15 ; lane L>=16 -> row M=L-16, K=k0+16..31
__device__ __forceinline__ v16h load_fragA_lds(const _Float16* __restrict__ srow,
                                               int kbase, int koff) {
    v8h a0 = *(const v8h*)(srow + kbase + koff);
    v8h a1 = *(const v8h*)(srow + kbase + koff + 8);
    return __builtin_shufflevector(a0, a1,
        0, 1, 2, 3, 4, 5, 6, 7, 8, 9, 10, 11, 12, 13, 14, 15);
}

__device__ __forceinline__ float pick8(v8f a, int i) {
    float r = a[0];
    r = (i == 1) ? a[1] : r;
    r = (i == 2) ? a[2] : r;
    r = (i == 3) ? a[3] : r;
    r = (i == 4) ? a[4] : r;
    r = (i == 5) ? a[5] : r;
    r = (i == 6) ? a[6] : r;
    r = (i == 7) ? a[7] : r;
    return r;
}

// ---------------------------------------------------------------------------
// Kernel G: fused WMMA GEMM + fixed-shift sum-exp + diagonal capture.
// Grid: (32 m-blocks, NCHUNK n-chunks, BATCH). Block: 256 threads = 8 waves.
// A tile (16x128 f16, 4KB contiguous) staged to LDS once per WG via async
// global->LDS copies (double-buffered; fill of tile i+1 overlaps compute of
// tile i), shared by all 8 waves. C initialized to -SHIFT so the epilogue is
// exp+add only and partial column sums combine across WGs with float atomics.
// ---------------------------------------------------------------------------
__global__ void __launch_bounds__(256)
fml_gemm_lse(const _Float16* __restrict__ An,
             const _Float16* __restrict__ Bn,
             float* __restrict__ sumexp,
             float* __restrict__ diagArr) {
    __shared__ __align__(16) _Float16 smem[2][16 * LDS_PITCH];

    const int tid  = threadIdx.x;
    const int lane = tid & 31;
    const int wave = tid >> 5;                       // 0..7
    const int b = blockIdx.z;
    const int nchunk = blockIdx.y;
    const int mbase = (blockIdx.x << 7) + (wave << 4);   // wave's 16 columns

    // Hoisted B fragments: 16 columns x K=128 (4 chunks of 32).
    const _Float16* colptr = Bn + ((size_t)(b * HW + mbase + (lane & 15))) * CHAN;
    const v16h bf0 = load_fragB(colptr, 0,  lane);
    const v16h bf1 = load_fragB(colptr, 32, lane);
    const v16h bf2 = load_fragB(colptr, 64, lane);
    const v16h bf3 = load_fragB(colptr, 96, lane);

    // A panel for this WG's n-chunk: rows are contiguous in packed layout.
    const _Float16* Ab = An + ((size_t)b * HW + (size_t)nchunk * NROWS) * CHAN;

    // Fill addressing: each thread moves one b128 (8 halfs) of the 4KB tile.
    const int frow = tid >> 4;          // 0..15
    const int fseg = tid & 15;          // 0..15 (8 halfs each)
    _Float16* fdst0 = &smem[0][frow * LDS_PITCH + fseg * 8];
    _Float16* fdst1 = &smem[1][frow * LDS_PITCH + fseg * 8];

    const _Float16* srowb = &smem[0][0] + (lane & 15) * LDS_PITCH;
    const int koff = (lane & 16) ? 16 : 0;
    const int sstride = 16 * LDS_PITCH;   // buffer stride in halfs

    float run_sum = 0.0f;
    float diag = 0.0f;

    // Prologue: fill tile 0 into buffer 0.
#if HAVE_ASYNC_LDS
    async_fill(Ab + tid * 8, fdst0);
#else
    {
        v8h g0 = *(const v8h*)(Ab + tid * 8);
        *(v8h*)fdst0 = g0;
    }
#endif
    wait_fill();
    __syncthreads();

    for (int i = 0; i < NTILES; ++i) {
        const _Float16* gnext = Ab + (size_t)(i + 1) * (16 * CHAN) + tid * 8;
        _Float16* sdst = (i & 1) ? fdst0 : fdst1;   // buffer (i+1)&1

        // Issue fill of the next tile (overlaps WMMA + exp work below).
#if HAVE_ASYNC_LDS
        if (i + 1 < NTILES) async_fill(gnext, sdst);
#else
        v8h g = {};
        if (i + 1 < NTILES) g = *(const v8h*)gnext;
#endif

        const _Float16* srow = srowb + (i & 1) * sstride;

        // Load all four A fragments into distinct registers so the ds_loads
        // can be claused and overlapped with the WMMA issue.
        v16h af0 = load_fragA_lds(srow, 0,  koff);
        v16h af1 = load_fragA_lds(srow, 32, koff);
        v16h af2 = load_fragA_lds(srow, 64, koff);
        v16h af3 = load_fragA_lds(srow, 96, koff);

        v8f acc = {-SHIFT, -SHIFT, -SHIFT, -SHIFT, -SHIFT, -SHIFT, -SHIFT, -SHIFT};
        acc = __builtin_amdgcn_wmma_f32_16x16x32_f16(false, af0, false, bf0,
                                                     (short)0, acc, false, false);
        acc = __builtin_amdgcn_wmma_f32_16x16x32_f16(false, af1, false, bf1,
                                                     (short)0, acc, false, false);
        acc = __builtin_amdgcn_wmma_f32_16x16x32_f16(false, af2, false, bf2,
                                                     (short)0, acc, false, false);
        acc = __builtin_amdgcn_wmma_f32_16x16x32_f16(false, af3, false, bf3,
                                                     (short)0, acc, false, false);
        // acc = sim[b, n0+M, mbase+N] - SHIFT   (tau folded into Bn)

        // Capture diagonal tile (wave-uniform condition, once per column).
        const int n0g = nchunk * NROWS + (i << 4);
        if (n0g == mbase) {
            int N = lane & 15;
            bool hi = lane >= 16;
            bool own = hi ? (N >= 8) : (N < 8);
            float d = pick8(acc, N & 7);
            diag = own ? d : 0.0f;
        }

        // Fixed-shift sum of exponentials (values <= 0, no overflow).
        float s = 0.0f;
        #pragma unroll
        for (int j = 0; j < 8; ++j) s += __expf(acc[j]);
        run_sum += s;

        // Commit staged data (fallback path), ensure fills landed, rendezvous.
#if !HAVE_ASYNC_LDS
        if (i + 1 < NTILES) *(v8h*)sdst = g;
#endif
        wait_fill();
        __syncthreads();
    }

    // Merge the half-columns held by lanes L and L+16.
    float S = run_sum + __shfl_xor(run_sum, 16, 32);
    float dfull = diag + __shfl_xor(diag, 16, 32);

    const bool ownsDiag = (mbase >= nchunk * NROWS) && (mbase < (nchunk + 1) * NROWS);
    if (lane < 16) {
        int idx = b * HW + mbase + lane;
        atomicAdd(&sumexp[idx], S);
        if (ownsDiag) diagArr[idx] = dfull;    // stores sim[k,k]-SHIFT
    }
}

// ---------------------------------------------------------------------------
// Kernel R: per-column loss = log(sum exp(sim-SHIFT)) - (diag-SHIFT)
//         = lse - diag  (SHIFT cancels). Masked, wave-reduced, atomics.
// ---------------------------------------------------------------------------
__global__ void fml_reduce(const float* __restrict__ sumexp,
                           const float* __restrict__ diagArr,
                           const float* __restrict__ mask,
                           float* __restrict__ accum) {
    int t = blockIdx.x * blockDim.x + threadIdx.x;   // 0 .. BATCH*HW-1
    float wm = mask[t];
    float loss = __logf(sumexp[t]) - diagArr[t];
    float num = loss * wm;
    float den = wm;
    #pragma unroll
    for (int off = 1; off < 32; off <<= 1) {
        num += __shfl_xor(num, off, 32);
        den += __shfl_xor(den, off, 32);
    }
    if ((threadIdx.x & 31) == 0) {
        atomicAdd(&accum[0], num);
        atomicAdd(&accum[1], den);
    }
}

// ---------------------------------------------------------------------------
// Kernel D: final scalar
// ---------------------------------------------------------------------------
__global__ void fml_finalize(const float* __restrict__ accum,
                             float* __restrict__ out) {
    if (threadIdx.x == 0 && blockIdx.x == 0)
        out[0] = accum[0] / (accum[1] + EPS_DEN);
}

// ---------------------------------------------------------------------------
// Launch: Z -> N -> G -> R -> D, all on `stream`.
// ws: [0,4MB) An | [4MB,8MB) Bn | +64KB sumexp | +64KB diag | +8B accum
// ---------------------------------------------------------------------------
extern "C" void kernel_launch(void* const* d_in, const int* in_sizes, int n_in,
                              void* d_out, int out_size, void* d_ws, size_t ws_size,
                              hipStream_t stream) {
    const float* featA = (const float*)d_in[0];
    const float* featB = (const float*)d_in[1];
    // d_in[2] = H_mat: unused by the reference computation.
    const float* mask  = (const float*)d_in[3];
    float* out = (float*)d_out;

    char* ws = (char*)d_ws;
    const size_t featHalfBytes = (size_t)BATCH * HW * CHAN * sizeof(_Float16); // 4 MB
    const size_t colBytes = (size_t)BATCH * HW * sizeof(float);                // 64 KB
    _Float16* An = (_Float16*)ws;
    _Float16* Bn = (_Float16*)(ws + featHalfBytes);
    float* sumexp  = (float*)(ws + 2 * featHalfBytes);
    float* diagArr = (float*)(ws + 2 * featHalfBytes + colBytes);
    float* accum   = (float*)(ws + 2 * featHalfBytes + 2 * colBytes);

    fml_zero<<<(BATCH * HW + 255) / 256, 256, 0, stream>>>(sumexp, accum);
    fml_normalize<<<(BATCH * HW) / 256, 256, 0, stream>>>(featA, featB, An, Bn);
    dim3 grid(HW / 128, NCHUNK, BATCH);   // (32, 4, 4) = 512 WGs x 8 waves
    fml_gemm_lse<<<grid, 256, 0, stream>>>(An, Bn, sumexp, diagArr);
    fml_reduce<<<(BATCH * HW) / 256, 256, 0, stream>>>(sumexp, diagArr, mask, accum);
    fml_finalize<<<1, 1, 0, stream>>>(accum, out);
}
